// InformerModel_15908558864593
// MI455X (gfx1250) — matrix operand order repaired and checked by reading, not compiled, not run on hardware
//
#include <hip/hip_runtime.h>
#include <math.h>

// ---------------- types for WMMA (gfx1250, wave32) ----------------
typedef __attribute__((ext_vector_type(16))) _Float16 v16h;
typedef __attribute__((ext_vector_type(8)))  float    v8f;

#define DM   512
#define DFF  2048
#define NH   8
#define DH   64
#define BSZ  4
#define LE   2048
#define LD   576
#define BH   (BSZ*NH)

// ==================================================================
// Embedding: out[b,l,d] = x[b,l,0]*w[d] + bias[d]
// ==================================================================
__global__ void embed_kernel(const float* __restrict__ x,
                             const float* __restrict__ w,
                             const float* __restrict__ b,
                             float* __restrict__ out, long total) {
  long i = (long)blockIdx.x * blockDim.x + threadIdx.x;
  if (i < total) {
    int d = (int)(i & (DM - 1));
    out[i] = x[i >> 9] * w[d] + b[d];
  }
}

// ==================================================================
// Tiled WMMA GEMM: C = act(A[M,K] @ W[K,N] + bias) (+residual)
// mode 0: bias only; mode 1: bias + exact GELU; mode 2: bias + residual
// Block: 128 threads = 4 waves; tile 64x64, K-step 32.
// LDS tiles are stored in WMMA *fragment order*: [tile][lane][16 halves],
// so each lane reads its whole fragment as one 32B (2 x ds_load_b128) load.
//   A layout (16-bit A 16x32): lane = (m&15) + 16*half, elem i <-> k where
//     k = (i<8 ? i : i+8) + 8*half
//   B layout (dense B 32x16): lane = (n&15) + 16*half, same elem i <-> k map as A
// ==================================================================
__global__ __launch_bounds__(128) void gemm_wmma_kernel(
    const float* __restrict__ A, const float* __restrict__ W,
    const float* __restrict__ bias, const float* __restrict__ res,
    float* __restrict__ C, int M, int N, int K, int mode) {
  __shared__ __align__(32) _Float16 Af[4][32][16];  // [m-tile][lane][elem]
  __shared__ __align__(32) _Float16 Bf[4][32][16];  // [n-tile][lane][elem]

  const int tid  = threadIdx.x;
  const int wave = tid >> 5;
  const int lane = tid & 31;
  const int lrow  = lane & 15;
  const int lhalf = lane >> 4;
  const int m0 = blockIdx.y * 64;
  const int n0 = blockIdx.x * 64;

  v8f acc[4] = {};

  for (int k0 = 0; k0 < K; k0 += 32) {
    __syncthreads();
    // ---- A tile: 64 rows x 32 k = 512 quads of 4 consecutive k ----
    for (int qa = tid; qa < 512; qa += 128) {
      const int r  = qa >> 3;          // row in tile 0..63
      const int kq = (qa & 7) << 2;    // k offset 0,4,...,28
      const float4 f = *reinterpret_cast<const float4*>(
          A + (size_t)(m0 + r) * K + (k0 + kq));
      const int half = (kq >> 3) & 1;
      const int i0   = (kq & 4) | ((kq & 16) >> 1);
      union { _Float16 h[4]; unsigned long long u; } pk;
      pk.h[0] = (_Float16)f.x; pk.h[1] = (_Float16)f.y;
      pk.h[2] = (_Float16)f.z; pk.h[3] = (_Float16)f.w;
      *reinterpret_cast<unsigned long long*>(
          &Af[r >> 4][(r & 15) + 16 * half][i0]) = pk.u;
    }
    // ---- B tile: 32 k x 64 n = 512 quads of 4 consecutive k ----
    for (int qb = tid; qb < 512; qb += 128) {
      const int n  = qb & 63;          // column in tile
      const int kq = (qb >> 6) << 2;   // k offset 0,4,...,28
      const float* wp = W + (size_t)(k0 + kq) * N + (n0 + n);
      union { _Float16 h[4]; unsigned long long u; } pk;
      pk.h[0] = (_Float16)wp[0];
      pk.h[1] = (_Float16)wp[(size_t)N];
      pk.h[2] = (_Float16)wp[(size_t)2 * N];
      pk.h[3] = (_Float16)wp[(size_t)3 * N];
      const int half = (kq >> 3) & 1;
      const int i0   = (kq & 4) | ((kq & 16) >> 1);
      *reinterpret_cast<unsigned long long*>(
          &Bf[n >> 4][(n & 15) + 16 * half][i0]) = pk.u;
    }
    __syncthreads();

    const v16h a = *reinterpret_cast<const v16h*>(&Af[wave][lane][0]);
#pragma unroll
    for (int nt = 0; nt < 4; nt++) {
      const v16h bf = *reinterpret_cast<const v16h*>(&Bf[nt][lane][0]);
      acc[nt] = __builtin_amdgcn_wmma_f32_16x16x32_f16(
          false, a, false, bf, (short)0, acc[nt], false, false);
    }
  }

  // C layout: lane n = lane%16, VGPR v -> m = v + 8*(lane/16)
#pragma unroll
  for (int nt = 0; nt < 4; nt++) {
#pragma unroll
    for (int v = 0; v < 8; v++) {
      int m = m0 + wave * 16 + v + 8 * lhalf;
      int n = n0 + nt * 16 + lrow;
      float val = acc[nt][v] + bias[n];
      if (mode == 1) {
        val = 0.5f * val * (1.0f + erff(val * 0.70710678118f));
      } else if (mode == 2) {
        val += res[(size_t)m * N + n];
      }
      C[(size_t)m * N + n] = val;
    }
  }
}

// ==================================================================
// LayerNorm over last dim (512), one block per row
// ==================================================================
__global__ __launch_bounds__(128) void ln_kernel(const float* __restrict__ x,
                                                 const float* __restrict__ g,
                                                 const float* __restrict__ bt,
                                                 float* __restrict__ out) {
  const int row = blockIdx.x;
  const int tid = threadIdx.x;
  const float* xr = x + (size_t)row * DM;
  float v[4];
  float s = 0.f;
#pragma unroll
  for (int i = 0; i < 4; i++) { v[i] = xr[tid + i * 128]; s += v[i]; }
  __shared__ float red[128];
  red[tid] = s; __syncthreads();
  for (int off = 64; off > 0; off >>= 1) {
    if (tid < off) red[tid] += red[tid + off];
    __syncthreads();
  }
  float mean = red[0] * (1.0f / DM);
  __syncthreads();
  float sv = 0.f;
#pragma unroll
  for (int i = 0; i < 4; i++) { float d = v[i] - mean; sv += d * d; }
  red[tid] = sv; __syncthreads();
  for (int off = 64; off > 0; off >>= 1) {
    if (tid < off) red[tid] += red[tid + off];
    __syncthreads();
  }
  float inv = rsqrtf(red[0] * (1.0f / DM) + 1e-5f);
#pragma unroll
  for (int i = 0; i < 4; i++) {
    int d = tid + i * 128;
    out[(size_t)row * DM + d] = (v[i] - mean) * inv * g[d] + bt[d];
  }
}

// ==================================================================
// ProbSparse attention pieces
// ==================================================================
__device__ __forceinline__ unsigned int hashu(unsigned int x) {
  x ^= x >> 16; x *= 0x7feb352dU; x ^= x >> 15; x *= 0x846ca68bU; x ^= x >> 16;
  return x;
}

// M[bh,l] = max_j(q.k_samp) - sum_j(q.k_samp)/Lk
__global__ __launch_bounds__(128) void mscore_kernel(
    const float* __restrict__ xq, const float* __restrict__ xkv,
    float* __restrict__ Mout, int Lq, int Lk, int U, unsigned int seed) {
  const int bh = blockIdx.y;
  const int b = bh >> 3, h = bh & 7;
  const int l = blockIdx.x * 128 + threadIdx.x;
  if (l >= Lq) return;
  const float* q = xq + ((size_t)b * Lq + l) * DM + h * DH;
  float qr[DH];
#pragma unroll
  for (int d = 0; d < DH; d++) qr[d] = q[d];
  float mx = -1e30f, sm = 0.f;
  for (int j = 0; j < U; j++) {
    int kidx = (int)(hashu(seed ^ (unsigned)(l * 131u + (unsigned)j * 2654435761u)) % (unsigned)Lk);
    const float* kp = xkv + ((size_t)b * Lk + kidx) * DM + h * DH;
    float dot = 0.f;
#pragma unroll
    for (int d = 0; d < DH; d++) dot += qr[d] * kp[d];
    mx = fmaxf(mx, dot);
    sm += dot;
  }
  Mout[(size_t)bh * Lq + l] = mx - sm / (float)Lk;
}

// top-u selection (destructive on Mv), one block per bh
__global__ __launch_bounds__(256) void topk_kernel(float* __restrict__ Mv,
                                                   int* __restrict__ mtop,
                                                   int Lq, int u) {
  const int bh = blockIdx.x;
  float* Mr = Mv + (size_t)bh * Lq;
  __shared__ float sval[256];
  __shared__ int   sidx[256];
  const int tid = threadIdx.x;
  for (int r = 0; r < u; r++) {
    float best = -1e30f; int bi = 0;
    for (int i = tid; i < Lq; i += 256) {
      float val = Mr[i];
      if (val > best) { best = val; bi = i; }
    }
    sval[tid] = best; sidx[tid] = bi; __syncthreads();
    for (int off = 128; off > 0; off >>= 1) {
      if (tid < off && sval[tid + off] > sval[tid]) {
        sval[tid] = sval[tid + off]; sidx[tid] = sidx[tid + off];
      }
      __syncthreads();
    }
    if (tid == 0) { mtop[bh * 64 + r] = sidx[0]; Mr[sidx[0]] = -1e30f; }
    __syncthreads();
  }
}

// vmean[bh,d] = mean over keys of v
__global__ __launch_bounds__(64) void vmean_kernel(const float* __restrict__ xkv,
                                                   float* __restrict__ vm, int Lk) {
  const int bh = blockIdx.x, b = bh >> 3, h = bh & 7, d = threadIdx.x;
  float s = 0.f;
  for (int l = 0; l < Lk; l++) s += xkv[((size_t)b * Lk + l) * DM + h * DH + d];
  vm[bh * DH + d] = s / (float)Lk;
}

// ctx[bh,l,d] = vmean[bh,d]
__global__ void ctxinit_kernel(const float* __restrict__ vm,
                               float* __restrict__ ctx, int Lq, long total) {
  long i = (long)blockIdx.x * blockDim.x + threadIdx.x;
  if (i < total) {
    int d = (int)(i & (DH - 1));
    int bh = (int)(i / ((long)Lq * DH));
    ctx[i] = vm[bh * DH + d];
  }
}

// full softmax update for selected queries; scatter into ctx
__global__ __launch_bounds__(128) void attn_update_kernel(
    const float* __restrict__ xq, const float* __restrict__ xkv,
    const int* __restrict__ mtop, float* __restrict__ ctx,
    int Lq, int Lk, int u) {
  const int bh = blockIdx.y, ui = blockIdx.x;
  const int b = bh >> 3, h = bh & 7;
  const int qrow = mtop[bh * 64 + ui];
  const int tid = threadIdx.x;
  __shared__ float qs[DH];
  __shared__ float sc[2048];
  __shared__ float red[128];
  if (tid < DH) qs[tid] = xq[((size_t)b * Lq + qrow) * DM + h * DH + tid];
  __syncthreads();
  float lmax = -1e30f;
  for (int k = tid; k < Lk; k += 128) {
    const float* kp = xkv + ((size_t)b * Lk + k) * DM + h * DH;
    float dot = 0.f;
#pragma unroll
    for (int d = 0; d < DH; d++) dot += qs[d] * kp[d];
    dot *= 0.125f;  // 1/sqrt(64)
    sc[k] = dot;
    lmax = fmaxf(lmax, dot);
  }
  red[tid] = lmax; __syncthreads();
  for (int off = 64; off > 0; off >>= 1) {
    if (tid < off) red[tid] = fmaxf(red[tid], red[tid + off]);
    __syncthreads();
  }
  float gmax = red[0]; __syncthreads();
  float lsum = 0.f;
  for (int k = tid; k < Lk; k += 128) {
    float e = __expf(sc[k] - gmax);
    sc[k] = e; lsum += e;
  }
  red[tid] = lsum; __syncthreads();
  for (int off = 64; off > 0; off >>= 1) {
    if (tid < off) red[tid] += red[tid + off];
    __syncthreads();
  }
  float inv = 1.0f / red[0];
  __syncthreads();
  if (tid < DH) {
    const int d = tid;
    float accv = 0.f;
    for (int k = 0; k < Lk; k++)
      accv += sc[k] * xkv[((size_t)b * Lk + k) * DM + h * DH + d];
    ctx[((size_t)bh * Lq + qrow) * DH + d] = accv * inv;
  }
}

// out[b,l,h*64+d] = x[b,l,h*64+d] + ctx[(b*8+h),l,d]
__global__ void merge_kernel(const float* __restrict__ x,
                             const float* __restrict__ ctx,
                             float* __restrict__ out, int L, long total) {
  long i = (long)blockIdx.x * blockDim.x + threadIdx.x;
  if (i < total) {
    int d = (int)(i & (DH - 1));
    int h = (int)((i >> 6) & 7);
    long bl = i >> 9;
    int b = (int)(bl / L), l = (int)(bl % L);
    out[i] = x[i] + ctx[(((size_t)(b * NH + h) * L + l) * DH) + d];
  }
}

// final projection: out[row] = x[row,:] . w + b0
__global__ __launch_bounds__(128) void proj_kernel(const float* __restrict__ x,
                                                   const float* __restrict__ w,
                                                   const float* __restrict__ b0,
                                                   float* __restrict__ out) {
  const int row = blockIdx.x, tid = threadIdx.x;
  float s = 0.f;
  for (int i = tid; i < DM; i += 128) s += x[(size_t)row * DM + i] * w[i];
  __shared__ float red[128];
  red[tid] = s; __syncthreads();
  for (int off = 64; off > 0; off >>= 1) {
    if (tid < off) red[tid] += red[tid + off];
    __syncthreads();
  }
  if (tid == 0) out[row] = red[0] + b0[0];
}

// ==================================================================
// Host orchestration
// ==================================================================
extern "C" void kernel_launch(void* const* d_in, const int* in_sizes, int n_in,
                              void* d_out, int out_size, void* d_ws, size_t ws_size,
                              hipStream_t stream) {
  (void)in_sizes; (void)n_in; (void)out_size; (void)ws_size;
  auto F = [&](int i) { return (const float*)d_in[i]; };

  // setup_inputs() dict-insertion order:
  // 0 x_enc, 1 x_mark_enc, 2 x_dec, 3 x_mark_dec, then params:
  // 4 enc_emb_w, 5 enc_emb_b, 6 dec_emb_w, 7 dec_emb_b,
  // 8..15  enc L0 {w1,b1,w2,b2,n1g,n1b,n2g,n2b}
  // 16..23 enc L1 {same}
  // 24..33 dec L0 {w1,b1,w2,b2,n1g,n1b,n2g,n2b,n3g,n3b}
  // 34 proj_w, 35 proj_b, 36 enc_ng, 37 enc_nb, 38 dec_ng, 39 dec_nb
  const float* x_enc = F(0);
  const float* x_dec = F(2);

  // workspace layout (floats)
  const size_t SE  = (size_t)BSZ * LE * DM;   // 4.19M
  const size_t SEF = (size_t)BSZ * LE * DFF;  // 16.8M
  const size_t SD  = (size_t)BSZ * LD * DM;   // 1.18M
  const size_t SDF = (size_t)BSZ * LD * DFF;  // 4.72M
  float* ws   = (float*)d_ws;
  float* x_e  = ws;            ws += SE;
  float* ctx_e = ws;           ws += SE;   // holds BH*Lq*DH
  float* t_e  = ws;            ws += SE;
  float* mid_e = ws;           ws += SEF;
  float* x_d  = ws;            ws += SD;
  float* ctx_d = ws;           ws += SD;
  float* t_d  = ws;            ws += SD;
  float* mid_d = ws;           ws += SDF;
  float* Mbuf = ws;            ws += (size_t)BH * LE;
  float* vmb  = ws;            ws += (size_t)BH * DH;
  int*   mtop = (int*)ws;      // BH*64 ints

  auto run_ln = [&](const float* x, const float* g, const float* b, float* o, int rows) {
    ln_kernel<<<rows, 128, 0, stream>>>(x, g, b, o);
  };

  auto run_attn = [&](const float* xq, const float* xkv, float* ctx, float* tout,
                      int Lq, int Lk, int U, int u, unsigned seed) {
    dim3 g1((unsigned)((Lq + 127) / 128), BH);
    mscore_kernel<<<g1, 128, 0, stream>>>(xq, xkv, Mbuf, Lq, Lk, U, seed);
    topk_kernel<<<BH, 256, 0, stream>>>(Mbuf, mtop, Lq, u);
    vmean_kernel<<<BH, 64, 0, stream>>>(xkv, vmb, Lk);
    long ctot = (long)BH * Lq * DH;
    ctxinit_kernel<<<(unsigned)((ctot + 255) / 256), 256, 0, stream>>>(vmb, ctx, Lq, ctot);
    dim3 g2((unsigned)u, BH);
    attn_update_kernel<<<g2, 128, 0, stream>>>(xq, xkv, mtop, ctx, Lq, Lk, u);
    long mtot = (long)BSZ * Lq * DM;
    merge_kernel<<<(unsigned)((mtot + 255) / 256), 256, 0, stream>>>(xq, ctx, tout, Lq, mtot);
  };

  auto run_gemm = [&](const float* A, const float* W, const float* bias,
                      const float* res, float* C, int M, int N, int K, int mode) {
    dim3 grid((unsigned)(N / 64), (unsigned)(M / 64));
    gemm_wmma_kernel<<<grid, 128, 0, stream>>>(A, W, bias, res, C, M, N, K, mode);
  };

  // -------- encoder --------
  embed_kernel<<<(unsigned)((SE + 255) / 256), 256, 0, stream>>>(x_enc, F(4), F(5), x_e, (long)SE);
  const int ME = BSZ * LE;  // 8192
  for (int l = 0; l < 2; l++) {
    const int base = 8 + l * 8;
    run_attn(x_e, x_e, ctx_e, t_e, LE, LE, 40, 40, 0x9e3779b9u + (unsigned)l);
    run_ln(t_e, F(base + 4), F(base + 5), x_e, ME);  // n1
    run_ln(x_e, F(base + 6), F(base + 7), x_e, ME);  // n2 (double LN quirk)
    run_gemm(x_e, F(base + 0), F(base + 1), nullptr, mid_e, ME, DFF, DM, 1);  // gelu
    run_gemm(mid_e, F(base + 2), F(base + 3), x_e, t_e, ME, DM, DFF, 2);      // +residual
    run_ln(t_e, F(base + 6), F(base + 7), x_e, ME);  // n2
  }
  run_ln(x_e, F(36), F(37), x_e, ME);  // enc final LN -> cross memory

  // -------- decoder --------
  embed_kernel<<<(unsigned)((SD + 255) / 256), 256, 0, stream>>>(x_dec, F(6), F(7), x_d, (long)SD);
  const int MD = BSZ * LD;  // 2304
  {
    const int base = 24;
    // self prob-attn (Lq=Lk=576 -> U=u=35)
    run_attn(x_d, x_d, ctx_d, t_d, LD, LD, 35, 35, 0x2545F491u);
    run_ln(t_d, F(base + 4), F(base + 5), x_d, MD);  // n1
    // cross prob-attn (Lq=576, Lk=2048 -> U=40, u=35)
    run_attn(x_d, x_e, ctx_d, t_d, LD, LE, 40, 35, 0x85EBCA77u);
    run_ln(t_d, F(base + 6), F(base + 7), x_d, MD);  // n2
    run_ln(x_d, F(base + 8), F(base + 9), x_d, MD);  // n3 (double LN quirk)
    run_gemm(x_d, F(base + 0), F(base + 1), nullptr, mid_d, MD, DFF, DM, 1);
    run_gemm(mid_d, F(base + 2), F(base + 3), x_d, t_d, MD, DM, DFF, 2);
    run_ln(t_d, F(base + 8), F(base + 9), x_d, MD);  // n3
  }
  run_ln(x_d, F(38), F(39), x_d, MD);  // dec final LN
  proj_kernel<<<MD, 128, 0, stream>>>(x_d, F(34), F(35), (float*)d_out);
}
